// GraphNet_35089882808442
// MI455X (gfx1250) — matrix operand order repaired and checked by reading, not compiled, not run on hardware
//
#include <hip/hip_runtime.h>
#include <hip/hip_bf16.h>

typedef __attribute__((ext_vector_type(16))) _Float16 v16h;
typedef __attribute__((ext_vector_type(8)))  float    v8f;

#define F_IN 512
#define HID  128
#define HEADS 8
#define CPH  16
#define MLPH 256
#define NEG_SLOPE 0.2f

// ---------------------------------------------------------------------------
// Swizzle a row-major f32 weight W[K, Ncols] into WMMA B-fragment order (f16):
// Wfrag[((kt*ntiles + nt)*32 + lane)*16 + i] = W[(kt*32 + (lane>>4)*8 + (i<8 ? i : i+8))*Ncols
//                                               + nt*16 + (lane&15)]
// so each lane's v16h B fragment is one contiguous 32-byte load.
__global__ void swizzle_b_kernel(const float* __restrict__ W, _Float16* __restrict__ Wfrag,
                                 int K, int Ncols) {
    const int ktiles = K >> 5, ntiles = Ncols >> 4;
    const int total = ktiles * ntiles * 32 * 16;
    int idx = blockIdx.x * blockDim.x + threadIdx.x;
    if (idx >= total) return;
    const int i    = idx & 15;
    const int lane = (idx >> 4) & 31;
    const int nt   = (idx >> 9) % ntiles;
    const int kt   = (idx >> 9) / ntiles;
    const int k    = kt * 32 + (lane >> 4) * 8 + (i < 8 ? i : i + 8);
    const int col  = nt * 16 + (lane & 15);
    Wfrag[idx] = (_Float16)W[(size_t)k * Ncols + col];
}

// zero-fill u32 words
__global__ void zero_u32_kernel(unsigned* __restrict__ p, size_t n) {
    size_t i = (size_t)blockIdx.x * blockDim.x + threadIdx.x;
    if (i < n) p[i] = 0u;
}

// ---------------------------------------------------------------------------
// Node-feature GEMM: Hout[N,128] = X[N,K] @ W[K,128]  (f16 WMMA, f32 accum)
// grid.x = ceil(N/16) blocks of 256 threads (8 waves); wave w -> 16-col tile w.
// Bfrag is pre-swizzled (see swizzle_b_kernel); ntiles = 8.
template <int K>
__global__ __launch_bounds__(256) void gemm_node_kernel(
    const float* __restrict__ X, const v16h* __restrict__ Bfrag,
    float* __restrict__ Hout, int nrows) {
    const int mtile = blockIdx.x;
    const int wave  = threadIdx.x >> 5;
    const int lane  = threadIdx.x & 31;
    const int row0  = mtile * 16;
    const int col0  = wave * 16;
    const int m     = lane & 15;
    const int half  = lane >> 4;      // 0: lanes 0-15, 1: lanes 16-31
    const int NT    = HID / 16;       // 8 n-tiles

    const int row = row0 + m;
    const float* xrow = X + (size_t)((row < nrows) ? row : 0) * K;

    v8f acc = {};
    for (int k0 = 0; k0 < K; k0 += 32) {
        // A fragment: 16x32 f16 (lanes<16: K {0..7,16..23}; lanes>=16: {8..15,24..31})
        const int kb = k0 + half * 8;
        const float4 x0 = *(const float4*)(xrow + kb);
        const float4 x1 = *(const float4*)(xrow + kb + 4);
        const float4 x2 = *(const float4*)(xrow + kb + 16);
        const float4 x3 = *(const float4*)(xrow + kb + 20);
        v16h a;
        a[0]=(_Float16)x0.x; a[1]=(_Float16)x0.y; a[2]=(_Float16)x0.z; a[3]=(_Float16)x0.w;
        a[4]=(_Float16)x1.x; a[5]=(_Float16)x1.y; a[6]=(_Float16)x1.z; a[7]=(_Float16)x1.w;
        a[8]=(_Float16)x2.x; a[9]=(_Float16)x2.y; a[10]=(_Float16)x2.z; a[11]=(_Float16)x2.w;
        a[12]=(_Float16)x3.x; a[13]=(_Float16)x3.y; a[14]=(_Float16)x3.z; a[15]=(_Float16)x3.w;
        // B fragment: single contiguous 32-byte load from swizzled weights
        const v16h b = Bfrag[(size_t)(((k0 >> 5) * NT + wave) * 32 + lane)];
        acc = __builtin_amdgcn_wmma_f32_16x16x32_f16(false, a, false, b, (short)0, acc,
                                                     false, false);
    }
    // D: col = col0 + (lane&15), rows v + 8*half
    const int col = col0 + m;
#pragma unroll
    for (int v = 0; v < 8; ++v) {
        const int r = row0 + v + 8 * half;
        if (r < nrows) Hout[(size_t)r * HID + col] = acc[v];
    }
}

// ---------------------------------------------------------------------------
// alpha_src / alpha_dst: per (node, head) 16-wide dot products
__global__ void alphas_kernel(const float* __restrict__ H,
                              const float* __restrict__ a_src, const float* __restrict__ a_dst,
                              float* __restrict__ as, float* __restrict__ ad, int N) {
    int i = blockIdx.x * blockDim.x + threadIdx.x;   // n*8 + h
    if (i >= N * HEADS) return;
    const int n = i >> 3, h = i & 7;
    const float* hr = H + (size_t)n * HID + h * CPH;
    float s = 0.f, d = 0.f;
#pragma unroll
    for (int c = 0; c < CPH; ++c) {
        s += hr[c] * a_src[h * CPH + c];
        d += hr[c] * a_dst[h * CPH + c];
    }
    as[i] = s;
    ad[i] = d;
}

__device__ __forceinline__ unsigned enc_f32(float f) {
    unsigned u = __float_as_uint(f);
    return (u & 0x80000000u) ? ~u : (u | 0x80000000u);
}
__device__ __forceinline__ float dec_f32(unsigned u) {
    return (u & 0x80000000u) ? __uint_as_float(u & 0x7FFFFFFFu) : __uint_as_float(~u);
}
__device__ __forceinline__ void edge_nodes(const int* ei, int e, int E, int& s, int& d) {
    if (e < E) { s = ei[e]; d = ei[E + e]; }
    else       { s = d = e - E; }            // appended self-loops
}

// segment max (encoded-uint atomicMax)
__global__ void seg_max_kernel(const int* __restrict__ ei, const float* __restrict__ as,
                               const float* __restrict__ ad, unsigned* __restrict__ menc,
                               int E, int N) {
    size_t i = (size_t)blockIdx.x * blockDim.x + threadIdx.x;
    if (i >= (size_t)(E + N) * HEADS) return;
    const int e = (int)(i >> 3), h = (int)(i & 7);
    int s, d; edge_nodes(ei, e, E, s, d);
    float x = as[s * HEADS + h] + ad[d * HEADS + h];
    x = x > 0.f ? x : NEG_SLOPE * x;
    atomicMax(&menc[d * HEADS + h], enc_f32(x));
}

// exp(e - max) + segment sum; stores per-edge exp
__global__ void seg_exp_kernel(const int* __restrict__ ei, const float* __restrict__ as,
                               const float* __restrict__ ad, const unsigned* __restrict__ menc,
                               float* __restrict__ eexp, float* __restrict__ ssum,
                               int E, int N) {
    size_t i = (size_t)blockIdx.x * blockDim.x + threadIdx.x;
    if (i >= (size_t)(E + N) * HEADS) return;
    const int e = (int)(i >> 3), h = (int)(i & 7);
    int s, d; edge_nodes(ei, e, E, s, d);
    float x = as[s * HEADS + h] + ad[d * HEADS + h];
    x = x > 0.f ? x : NEG_SLOPE * x;
    const float mx = dec_f32(menc[d * HEADS + h]);
    const float ex = expf(x - mx);
    eexp[i] = ex;
    atomicAdd(&ssum[d * HEADS + h], ex);
}

// alpha-weighted aggregation: Out[dst, h*16+c] += alpha * Hlin[src, h*16+c]
__global__ void seg_agg_kernel(const int* __restrict__ ei, const float* __restrict__ Hlin,
                               const float* __restrict__ eexp, const float* __restrict__ ssum,
                               float* __restrict__ Out, int E, int N) {
    size_t i = (size_t)blockIdx.x * blockDim.x + threadIdx.x;
    if (i >= (size_t)(E + N) * HEADS) return;
    const int e = (int)(i >> 3), h = (int)(i & 7);
    int s, d; edge_nodes(ei, e, E, s, d);
    const float alpha = eexp[i] / ssum[d * HEADS + h];
    const float* hs = Hlin + (size_t)s * HID + h * CPH;
    float* o = Out + (size_t)d * HID + h * CPH;
#pragma unroll
    for (int c = 0; c < CPH; ++c) atomicAdd(&o[c], alpha * hs[c]);
}

// out = relu(out + bias[col])
__global__ void bias_relu_kernel(float* __restrict__ Out, const float* __restrict__ b, size_t n) {
    size_t i = (size_t)blockIdx.x * blockDim.x + threadIdx.x;
    if (i >= n) return;
    float v = Out[i] + b[i & (HID - 1)];
    Out[i] = v > 0.f ? v : 0.f;
}

// ---------------------------------------------------------------------------
// Edge MLP: 16 edges per block; hid = relu([h[src]||h[dst]] @ W1 + b1); pred = hid@w2 + b2
// Bfrag pre-swizzled: ntiles = 16, ktiles = 8.
__global__ __launch_bounds__(256) void edge_mlp_kernel(
    const float* __restrict__ Hf, const int* __restrict__ ei,
    const v16h* __restrict__ Bfrag, const float* __restrict__ b1,
    const float* __restrict__ w2, const float* __restrict__ b2,
    float* __restrict__ pred, int E) {
    __shared__ _Float16 Ah[16 * MLPH];   // 8 KB gathered A tile (f16)
    __shared__ float    Hid[16 * MLPH];  // 16 KB hidden activations
    __shared__ float    Part[16 * 16];   // reduction partials

    const int eb = blockIdx.x * 16;
    const int t  = threadIdx.x;

    // stage gathered edge features into LDS as f16
    {
        const int e = t >> 4;               // edge within block
        const int c0 = (t & 15) * 16;       // 16-col segment
        const int ge = eb + e;
        const float* srow;
        const float* drow;
        if (ge < E) {
            srow = Hf + (size_t)ei[ge] * HID;
            drow = Hf + (size_t)ei[E + ge] * HID;
        } else { srow = Hf; drow = Hf; }
#pragma unroll
        for (int i = 0; i < 16; ++i) {
            const int c = c0 + i;
            const float v = (c < HID) ? srow[c] : drow[c - HID];
            Ah[e * MLPH + c] = (_Float16)v;
        }
    }
    __syncthreads();

    const int wave = t >> 5, lane = t & 31;
    const int m = lane & 15, half = lane >> 4;
    const int nt0 = wave * 2;               // this wave's two 16-col tiles
    const int NT = 16;

    v8f acc0 = {}, acc1 = {};
    for (int k0 = 0; k0 < MLPH; k0 += 32) {
        // A fragment from LDS: two contiguous 16-byte runs per lane
        const int kb = k0 + half * 8;
        v16h a;
        *(float4*)&a      = *(const float4*)&Ah[m * MLPH + kb];
        *((float4*)&a + 1) = *(const float4*)&Ah[m * MLPH + kb + 16];
        const int kt = k0 >> 5;
        const v16h bA = Bfrag[(size_t)((kt * NT + nt0) * 32 + lane)];
        const v16h bB = Bfrag[(size_t)((kt * NT + nt0 + 1) * 32 + lane)];
        acc0 = __builtin_amdgcn_wmma_f32_16x16x32_f16(false, a, false, bA, (short)0, acc0,
                                                      false, false);
        acc1 = __builtin_amdgcn_wmma_f32_16x16x32_f16(false, a, false, bB, (short)0, acc1,
                                                      false, false);
    }
    // bias + relu epilogue into LDS
#pragma unroll
    for (int v = 0; v < 8; ++v) {
        const int r  = v + 8 * half;
        const int c0 = nt0 * 16 + m, c1 = c0 + 16;
        float h0 = acc0[v] + b1[c0];
        float h1 = acc1[v] + b1[c1];
        Hid[r * MLPH + c0] = h0 > 0.f ? h0 : 0.f;
        Hid[r * MLPH + c1] = h1 > 0.f ? h1 : 0.f;
    }
    __syncthreads();

    // pred = hid @ w2 + b2 : 16 threads/edge partial dots, then 16-way sums
    {
        const int e = t >> 4, j = t & 15;
        float p = 0.f;
#pragma unroll
        for (int i = 0; i < 16; ++i) p += Hid[e * MLPH + j * 16 + i] * w2[j * 16 + i];
        Part[e * 16 + j] = p;
    }
    __syncthreads();
    if (t < 16) {
        const int ge = eb + t;
        if (ge < E) {
            float s = b2[0];
#pragma unroll
            for (int j = 0; j < 16; ++j) s += Part[t * 16 + j];
            pred[ge] = s;
        }
    }
}

// ---------------------------------------------------------------------------
extern "C" void kernel_launch(void* const* d_in, const int* in_sizes, int n_in,
                              void* d_out, int out_size, void* d_ws, size_t ws_size,
                              hipStream_t stream) {
    const float* x      = (const float*)d_in[0];
    const int*   ei     = (const int*)d_in[1];
    const float* W1     = (const float*)d_in[2];
    const float* a_src1 = (const float*)d_in[3];
    const float* a_dst1 = (const float*)d_in[4];
    const float* b1     = (const float*)d_in[5];
    const float* W2     = (const float*)d_in[6];
    const float* a_src2 = (const float*)d_in[7];
    const float* a_dst2 = (const float*)d_in[8];
    const float* b2     = (const float*)d_in[9];
    const float* mw1    = (const float*)d_in[10];
    const float* mb1    = (const float*)d_in[11];
    const float* mw2    = (const float*)d_in[12];
    const float* mb2    = (const float*)d_in[13];
    float* pred = (float*)d_out;

    const int N = in_sizes[0] / F_IN;
    const int E = in_sizes[1] / 2;

    // workspace carve (bump allocator, 256B aligned)
    char* p = (char*)d_ws;
    auto alloc = [&](size_t bytes) -> void* {
        void* r = (void*)p;
        p += (bytes + 255) & ~(size_t)255;
        return r;
    };
    v16h* W1f  = (v16h*)alloc((size_t)F_IN * HID * 2);        // swizzled f16
    v16h* W2f  = (v16h*)alloc((size_t)HID * HID * 2);
    v16h* MW1f = (v16h*)alloc((size_t)(2 * HID) * MLPH * 2);
    float* bufA = (float*)alloc((size_t)N * HID * 4);   // linear GEMM output
    float* bufB = (float*)alloc((size_t)N * HID * 4);   // aggregated output
    float* as   = (float*)alloc((size_t)N * HEADS * 4);
    float* ad   = (float*)alloc((size_t)N * HEADS * 4);
    unsigned* menc = (unsigned*)alloc((size_t)N * HEADS * 4);
    float* ssum = (float*)alloc((size_t)N * HEADS * 4);
    float* eexp = (float*)alloc((size_t)(E + N) * HEADS * 4);

    const int TPB = 256;
    const int mtiles = (N + 15) / 16;
    const size_t nh = (size_t)N * HEADS;
    const size_t nfeat = (size_t)N * HID;
    const size_t eh = (size_t)(E + N) * HEADS;
    const int nh_g   = (int)((nh + TPB - 1) / TPB);
    const int feat_g = (int)((nfeat + TPB - 1) / TPB);
    const int eh_g   = (int)((eh + TPB - 1) / TPB);

    // weight swizzles into WMMA fragment order
    swizzle_b_kernel<<<(F_IN * HID + TPB - 1) / TPB, TPB, 0, stream>>>(W1, (_Float16*)W1f, F_IN, HID);
    swizzle_b_kernel<<<(HID * HID + TPB - 1) / TPB, TPB, 0, stream>>>(W2, (_Float16*)W2f, HID, HID);
    swizzle_b_kernel<<<(2 * HID * MLPH + TPB - 1) / TPB, TPB, 0, stream>>>(mw1, (_Float16*)MW1f,
                                                                           2 * HID, MLPH);

    // ---- GAT layer 1 ----
    gemm_node_kernel<F_IN><<<mtiles, TPB, 0, stream>>>(x, W1f, bufA, N);
    alphas_kernel<<<nh_g, TPB, 0, stream>>>(bufA, a_src1, a_dst1, as, ad, N);
    zero_u32_kernel<<<nh_g, TPB, 0, stream>>>(menc, nh);
    zero_u32_kernel<<<nh_g, TPB, 0, stream>>>((unsigned*)ssum, nh);
    zero_u32_kernel<<<feat_g, TPB, 0, stream>>>((unsigned*)bufB, nfeat);
    seg_max_kernel<<<eh_g, TPB, 0, stream>>>(ei, as, ad, menc, E, N);
    seg_exp_kernel<<<eh_g, TPB, 0, stream>>>(ei, as, ad, menc, eexp, ssum, E, N);
    seg_agg_kernel<<<eh_g, TPB, 0, stream>>>(ei, bufA, eexp, ssum, bufB, E, N);
    bias_relu_kernel<<<feat_g, TPB, 0, stream>>>(bufB, b1, nfeat);

    // ---- GAT layer 2 ----
    gemm_node_kernel<HID><<<mtiles, TPB, 0, stream>>>(bufB, W2f, bufA, N);
    alphas_kernel<<<nh_g, TPB, 0, stream>>>(bufA, a_src2, a_dst2, as, ad, N);
    zero_u32_kernel<<<nh_g, TPB, 0, stream>>>(menc, nh);
    zero_u32_kernel<<<nh_g, TPB, 0, stream>>>((unsigned*)ssum, nh);
    zero_u32_kernel<<<feat_g, TPB, 0, stream>>>((unsigned*)bufB, nfeat);
    seg_max_kernel<<<eh_g, TPB, 0, stream>>>(ei, as, ad, menc, E, N);
    seg_exp_kernel<<<eh_g, TPB, 0, stream>>>(ei, as, ad, menc, eexp, ssum, E, N);
    seg_agg_kernel<<<eh_g, TPB, 0, stream>>>(ei, bufA, eexp, ssum, bufB, E, N);
    bias_relu_kernel<<<feat_g, TPB, 0, stream>>>(bufB, b2, nfeat);

    // ---- edge MLP ----
    edge_mlp_kernel<<<(E + 15) / 16, TPB, 0, stream>>>(bufB, ei, MW1f, mb1, mw2, mb2, pred, E);
}
